// AdaRRD_Decoder_87290915324643
// MI455X (gfx1250) — matrix operand order repaired and checked by reading, not compiled, not run on hardware
//
#include <hip/hip_runtime.h>
#include <math.h>

#define N_VAR   4096
#define M_CHK   2048
#define DV      4
#define DC      8
#define E_EDGES (N_VAR * DV)
#define BATCH   128
#define T_RRD   10
#define T_INNER 2
#define LLR_CLIP 15.0f
#define AMP_MIN_F 6.1180358e-07f  /* -log(tanh(15/2)) */

typedef __attribute__((ext_vector_type(16))) _Float16 v16h;
typedef __attribute__((ext_vector_type(8)))  float    v8f;

union F4 { float4 v; float f[4]; };

__device__ __forceinline__ float fast_rcp(float x) { return __builtin_amdgcn_rcpf(x); }

__device__ __forceinline__ float4 ld4(const float* p, int row, int lane) {
  return ((const float4*)(p + (size_t)row * BATCH))[lane];
}
__device__ __forceinline__ void st4(float* p, int row, int lane, float4 v) {
  ((float4*)(p + (size_t)row * BATCH))[lane] = v;
}

// ---------------------------------------------------------------------------
// Per-batch SNR estimate: E = mean_n chn^2 ; snr = 10*log10(E/((1+sqrt(1+E))*2))
// ---------------------------------------------------------------------------
__global__ void snr_kernel(const float* __restrict__ chn, float* __restrict__ snr) {
  int b = threadIdx.x;
  float acc = 0.0f;
  for (int n = 0; n < N_VAR; ++n) {
    float v = chn[n * BATCH + b];
    acc = fmaf(v, v, acc);
  }
  float E = acc * (1.0f / (float)N_VAR);
  float val = E * fast_rcp((1.0f + sqrtf(1.0f + E)) * 2.0f);  // 4*RATE = 2
  snr[b] = 10.0f * __log10f(val);
}

// ---------------------------------------------------------------------------
// Adapter MLP with WMMA.  Weights staged into zero-padded LDS (branch-free
// fragment packing, EXEC stays all-ones through the WMMAs).  Layer2
// (128x20 @ 20x4) = 8 tiles of v_wmma_f32_16x16x32_f16, K padded 20->32,
// N padded 4->16.  One wave.
// params layout: [beta(128) | gamma(128) | Wi(128) | We(128)]
// ---------------------------------------------------------------------------
__global__ void mlp_wmma_kernel(const float* __restrict__ snr,
                                const float* __restrict__ W1,
                                const float* __restrict__ b1,
                                const float* __restrict__ W2,
                                const float* __restrict__ b2,
                                float* __restrict__ params) {
  __shared__ float sW1[32];
  __shared__ float sb1[32];
  __shared__ float sW2[16][32];   // [n][k], zero padded
  __shared__ float sb2[16];

  const int L    = threadIdx.x;   // 0..31
  const int half = L >> 4;
  const int l15  = L & 15;

  // zero-pad + fill weights in LDS
  sW1[L] = 0.0f; sb1[L] = 0.0f;
  if (L < 16) sb2[L] = 0.0f;
  for (int i = L; i < 16 * 32; i += 32) (&sW2[0][0])[i] = 0.0f;
  __syncthreads();
  if (L < 20) { sW1[L] = W1[L]; sb1[L] = b1[L]; }
  if (L < 4)  sb2[L] = b2[L];
  for (int i = L; i < 80; i += 32) sW2[i / 20][i % 20] = W2[i];
  __syncthreads();

  // B fragment = W2^T (32K x 16N), 16-bit layout: half j -> K = (j<8 ? half*8+j : 16+half*8+j-8), N = lane&15
  v16h bm;
  const int n = l15;
  #pragma unroll
  for (int j = 0; j < 16; ++j) bm[j] = (_Float16)sW2[n][(j < 8) ? (half * 8 + j) : (16 + half * 8 + (j - 8))];

  #pragma unroll
  for (int m = 0; m < 8; ++m) {
    // A fragment = h tile (16M x 32K):
    // lanes 0-15 : row=16m+lane,    halves 0-7 -> K0-7,  8-15 -> K16-23
    // lanes 16-31: row=16m+lane-16, halves 0-7 -> K8-15, 8-15 -> K24-31
    int row = m * 16 + l15;
    float s = snr[row];
    v16h am;
    int k0 = half * 8;
    #pragma unroll
    for (int j = 0; j < 8; ++j) {
      int ka = k0 + j, kb = 16 + k0 + j;
      am[j]     = (_Float16)fmaxf(fmaf(s, sW1[ka], sb1[ka]), 0.0f);
      am[8 + j] = (_Float16)fmaxf(fmaf(s, sW1[kb], sb1[kb]), 0.0f);
    }

    v8f c = {};
    c = __builtin_amdgcn_wmma_f32_16x16x32_f16(
        false, am, false, bm, (short)0, c, false, false);

    // D layout: VGPR r -> M = 16m + r + 8*half, N = lane&15
    if (n < 4) {
      float scale = (n >= 2) ? 1.5f : 1.0f;  // Wi, We scaled by 1.5
      #pragma unroll
      for (int r = 0; r < 8; ++r) {
        int rr = m * 16 + r + half * 8;
        float x = c[r] + sb2[n];
        float sig = fast_rcp(1.0f + __expf(-x));
        params[n * BATCH + rr] = scale * sig;
      }
    }
  }
}

// ---------------------------------------------------------------------------
// Deterministic check->edge inverse index (one thread per check, sequential
// scan of row_idx: uniform-address broadcast loads, built once per launch).
// ---------------------------------------------------------------------------
__global__ void build_inv_kernel(const int* __restrict__ row_idx,
                                 int* __restrict__ chk_edges) {
  int m = blockIdx.x * blockDim.x + threadIdx.x;
  if (m >= M_CHK) return;
  int slot = 0;
  for (int e = 0; e < E_EDGES; ++e) {
    if (row_idx[e] == m) {
      chk_edges[m * DC + slot] = e;
      if (++slot == DC) break;
    }
  }
}

// ---------------------------------------------------------------------------
// Variable-node kernel, float4-vectorized over batch (32 lanes x 4 floats =
// one 512B row per wave, b128 loads/stores).  Block = 128 threads = 4 vars.
// Optionally builds soft_input, emits M_step (+carry), updates msg_V2C.
// ---------------------------------------------------------------------------
__global__ void var_kernel(const float* __restrict__ chn,
                           const float* __restrict__ prev,        // chn (t==0) or soft_output
                           const int*   __restrict__ perm,        // perms[t]
                           const float* __restrict__ params,
                           float*       __restrict__ soft_input,
                           const float* __restrict__ msg_C2V,
                           float*       __restrict__ msg_V2C,
                           float*       __restrict__ out_slice,
                           float*       __restrict__ soft_output,
                           int build_soft, int first_inner, int do_m, int do_v) {
  const int lane = threadIdx.x & 31;
  const int v    = blockIdx.x * 4 + (threadIdx.x >> 5);
  const float4* par4 = (const float4*)params;

  F4 be, ga, wi, we;
  be.v = par4[lane];
  ga.v = par4[32 + lane];
  wi.v = par4[64 + lane];
  we.v = par4[96 + lane];

  int pv = perm[v];

  F4 ell;
  if (build_soft) {
    F4 c, p;
    c.v = ld4(chn, pv, lane);
    p.v = ld4(prev, pv, lane);
    #pragma unroll
    for (int k = 0; k < 4; ++k)
      ell.f[k] = (1.0f - be.f[k]) * c.f[k] + be.f[k] * p.f[k];
    st4(soft_input, v, lane, ell.v);
  } else {
    ell.v = ld4(soft_input, v, lane);
  }

  F4 c2v[DV];
  F4 csum;
  #pragma unroll
  for (int k = 0; k < 4; ++k) csum.f[k] = 0.0f;
  if (!first_inner) {
    #pragma unroll
    for (int j = 0; j < DV; ++j) {
      c2v[j].v = ld4(msg_C2V, v * DV + j, lane);
      #pragma unroll
      for (int k = 0; k < 4; ++k) csum.f[k] += c2v[j].f[k];
    }
  }

  if (do_m) {
    F4 mv;
    #pragma unroll
    for (int k = 0; k < 4; ++k)
      mv.f[k] = wi.f[k] * ell.f[k] + (first_inner ? 0.0f : we.f[k] * csum.f[k]);
    st4(out_slice, pv, lane, mv.v);              // outputs[perm[v]] = M[v]
    if (soft_output) st4(soft_output, pv, lane, mv.v);
  }

  if (do_v) {
    #pragma unroll
    for (int j = 0; j < DV; ++j) {
      F4 old;
      if (!first_inner) old.v = ld4(msg_V2C, v * DV + j, lane);
      F4 nv;
      #pragma unroll
      for (int k = 0; k < 4; ++k) {
        float lam2 = first_inner ? 0.0f : we.f[k] * c2v[j].f[k];
        float tgt  = wi.f[k] * ell.f[k] +
                     (first_inner ? 0.0f : (we.f[k] * csum.f[k] - lam2));
        float ov   = first_inner ? 0.0f : old.f[k];
        nv.f[k] = (1.0f - ga.f[k]) * ov + ga.f[k] * tgt;
      }
      st4(msg_V2C, v * DV + j, lane, nv.v);
    }
  }
}

// ---------------------------------------------------------------------------
// Check-node kernel, float4-vectorized.  Block = 128 threads = 4 checks.
// Leave-one-out parity + log-tanh sums over the 8 edges of each check.
// log(tanh(a/2)) = log((1-e^-a)/(1+e^-a)); 2*atanh(x) = log((1+x)/(1-x)).
// Hardware transcendentals: v_exp_f32 / v_log_f32 / v_rcp_f32.
// ---------------------------------------------------------------------------
__global__ void chk_kernel(const int* __restrict__ chk_edges,
                           const float* __restrict__ params,
                           const float* __restrict__ msg_V2C,
                           float* __restrict__ msg_C2V,
                           int first_inner) {
  const int lane = threadIdx.x & 31;
  const int m    = blockIdx.x * 4 + (threadIdx.x >> 5);
  const float4* par4 = (const float4*)params;
  F4 ga; ga.v = par4[32 + lane];

  int   e[DC];
  float lt[DC * 4];
  int   neg[DC * 4];
  int   sneg[4] = {0, 0, 0, 0};
  float slt[4]  = {0.0f, 0.0f, 0.0f, 0.0f};

  #pragma unroll
  for (int j = 0; j < DC; ++j) {
    e[j] = chk_edges[m * DC + j];
    F4 l4; l4.v = ld4(msg_V2C, e[j], lane);
    #pragma unroll
    for (int c = 0; c < 4; ++c) {
      float lam = fminf(fmaxf(l4.f[c], -LLR_CLIP), LLR_CLIP);
      int ng = (lam < 0.0f) ? 1 : 0;
      float a = fminf(fmaxf(fabsf(lam), AMP_MIN_F), LLR_CLIP);
      float t = __expf(-a);
      float l = __logf((1.0f - t) * fast_rcp(1.0f + t));  // log(tanh(a/2))
      neg[j * 4 + c] = ng;  sneg[c] += ng;
      lt[j * 4 + c]  = l;   slt[c]  += l;
    }
  }

  #pragma unroll
  for (int j = 0; j < DC; ++j) {
    F4 old;
    if (!first_inner) old.v = ld4(msg_C2V, e[j], lane);
    F4 r;
    #pragma unroll
    for (int c = 0; c < 4; ++c) {
      int s = sneg[c] - neg[j * 4 + c];
      float sgn = (s & 1) ? -1.0f : 1.0f;
      float x = __expf(slt[c] - lt[j * 4 + c]) * (1.0f - 1e-6f);
      float o = sgn * __logf((1.0f + x) * fast_rcp(1.0f - x));  // sgn*2*atanh(x)
      float ov = first_inner ? 0.0f : old.f[c];
      r.f[c] = (1.0f - ga.f[c]) * ov + ga.f[c] * o;
    }
    st4(msg_C2V, e[j], lane, r.v);
  }
}

// ---------------------------------------------------------------------------
extern "C" void kernel_launch(void* const* d_in, const int* in_sizes, int n_in,
                              void* d_out, int out_size, void* d_ws, size_t ws_size,
                              hipStream_t stream) {
  (void)in_sizes; (void)n_in; (void)out_size; (void)ws_size;
  const float* chn     = (const float*)d_in[0];
  const float* W1      = (const float*)d_in[1];
  const float* b1      = (const float*)d_in[2];
  const float* W2      = (const float*)d_in[3];
  const float* b2      = (const float*)d_in[4];
  const int*   row_idx = (const int*)  d_in[5];
  /* d_in[6] = col_idx : structurally repeat(arange(N_VAR),4), not needed */
  const int*   perms   = (const int*)  d_in[7];
  /* d_in[8] = inv_perms : we scatter with perm instead */

  float* out = (float*)d_out;
  float* ws  = (float*)d_ws;

  // workspace layout (floats); all row bases 16B-aligned
  float* snr         = ws;                       // 128
  float* params      = ws + 128;                 // 512
  int*   chk_edges   = (int*)(ws + 1024);        // M_CHK*DC = 16384 ints
  float* soft_input  = ws + 1024 + M_CHK * DC;   // N*B
  float* soft_output = soft_input + (size_t)N_VAR * BATCH;
  float* msg_V2C     = soft_output + (size_t)N_VAR * BATCH;
  float* msg_C2V     = msg_V2C + (size_t)E_EDGES * BATCH;

  snr_kernel      <<<1, BATCH, 0, stream>>>(chn, snr);
  mlp_wmma_kernel <<<1, 32,    0, stream>>>(snr, W1, b1, W2, b2, params);
  build_inv_kernel<<<M_CHK / 128, 128, 0, stream>>>(row_idx, chk_edges);

  for (int t = 0; t < T_RRD; ++t) {
    const int*   perm = perms + (size_t)t * N_VAR;
    const float* prev = (t == 0) ? chn : soft_output;
    float* slice0 = out + (size_t)(t * T_INNER + 0) * N_VAR * BATCH;
    float* slice1 = out + (size_t)(t * T_INNER + 1) * N_VAR * BATCH;

    // inner iter 0: V (build soft_input; messages implicitly zero)
    var_kernel<<<N_VAR / 4, 128, 0, stream>>>(chn, prev, perm, params, soft_input,
                                              msg_C2V, msg_V2C, nullptr, nullptr,
                                              1, 1, 0, 1);
    chk_kernel<<<M_CHK / 4, 128, 0, stream>>>(chk_edges, params, msg_V2C, msg_C2V, 1);
    // M(iter0) fused with V(iter1): both per-var over the same C2V rows
    var_kernel<<<N_VAR / 4, 128, 0, stream>>>(chn, prev, perm, params, soft_input,
                                              msg_C2V, msg_V2C, slice0, nullptr,
                                              0, 0, 1, 1);
    chk_kernel<<<M_CHK / 4, 128, 0, stream>>>(chk_edges, params, msg_V2C, msg_C2V, 0);
    // final M: emit output + carry soft_output
    var_kernel<<<N_VAR / 4, 128, 0, stream>>>(chn, prev, perm, params, soft_input,
                                              msg_C2V, msg_V2C, slice1, soft_output,
                                              0, 0, 1, 0);
  }
}
